// PointerTransformer_18270790877520
// MI455X (gfx1250) — matrix operand order repaired and checked by reading, not compiled, not run on hardware
//
#include <hip/hip_runtime.h>
#include <math.h>

// ---------------------------------------------------------------------------
// Types for CDNA5 WMMA (wave32): V_WMMA_F32_16X16X32_BF16
// ---------------------------------------------------------------------------
typedef __bf16 bf16;
typedef bf16  bf16x8  __attribute__((ext_vector_type(8)));
typedef bf16  bf16x16 __attribute__((ext_vector_type(16)));
typedef float f32x8   __attribute__((ext_vector_type(8)));

#define WMMA_BF16(a, b, c) \
    __builtin_amdgcn_wmma_f32_16x16x32_bf16(false, (a), false, (b), (short)0, (c), false, false)

static __device__ __forceinline__ float wave_sum(float v) {
#pragma unroll
    for (int o = 16; o > 0; o >>= 1) v += __shfl_xor(v, o, 32);
    return v;
}
static __device__ __forceinline__ float wave_max(float v) {
#pragma unroll
    for (int o = 16; o > 0; o >>= 1) v = fmaxf(v, __shfl_xor(v, o, 32));
    return v;
}

// A fragment: 16(M) x 32(K) bf16, row-major source with leading dim lda.
// Lanes 0-15: M=lane, K = k0+0..15 contiguous
// Lanes 16-31: M=lane-16, K = k0+16..31 contiguous
static __device__ __forceinline__ bf16x16
load_a_frag(const bf16* __restrict__ A, long lda, int m0, int k0, int lane) {
    int half = lane >> 4;
    const bf16* r = A + (long)(m0 + (lane & 15)) * lda + k0 + half * 16;
    bf16x8 lo = *(const bf16x8*)(r);
    bf16x8 hi = *(const bf16x8*)(r + 8);
    bf16x16 a;
#pragma unroll
    for (int i = 0; i < 8; i++) { a[i] = lo[i]; a[i + 8] = hi[i]; }
    return a;
}

// B fragment: 32(K) x 16(N); source is W (N x K row-major, leading dim ldw),
// i.e. column n of B == row n of W.  Lane n (0-15): K=k0..k0+15 contiguous;
// lanes 16-31: K=k0+16..k0+31.
static __device__ __forceinline__ bf16x16
load_b_frag(const bf16* __restrict__ W, long ldw, int n0, int k0, int lane) {
    int half = lane >> 4;
    const bf16* r = W + (long)(n0 + (lane & 15)) * ldw + k0 + half * 16;
    return *(const bf16x16*)(r);
}

// ---------------------------------------------------------------------------
// f32 -> bf16 conversion
// ---------------------------------------------------------------------------
__global__ void cvt_kernel(const float* __restrict__ s, bf16* __restrict__ d, long n) {
    long i = (long)blockIdx.x * blockDim.x + threadIdx.x;
    if (i < n) d[i] = (bf16)s[i];
}

// ---------------------------------------------------------------------------
// Generic GEMM: C[m,n] = sum_k A[m,k] * W[n,k] + bias[n]   (dense(x,p) = x@w.T+b)
// A: M x K bf16 row-major. W: N x K bf16 row-major. One wave per 32x32 block
// (2x2 register-blocked WMMA tiles: 2 A-frags + 2 B-frags feed 4 WMMAs).
// Requires M%32==0, N%32==0, K%32==0.
// ---------------------------------------------------------------------------
__global__ void gemm_bf16_kernel(const bf16* __restrict__ A, const bf16* __restrict__ W,
                                 const float* __restrict__ bias,
                                 float* __restrict__ Cf, bf16* __restrict__ Cb,
                                 int M, int N, int K, int relu) {
    int wave = threadIdx.x >> 5, lane = threadIdx.x & 31;
    int n2 = N >> 5;
    int tile = blockIdx.x * (blockDim.x >> 5) + wave;
    if (tile >= (M >> 5) * n2) return;
    int m0 = (tile / n2) << 5, n0 = (tile % n2) << 5;
    f32x8 c00 = {}, c01 = {}, c10 = {}, c11 = {};
    for (int k0 = 0; k0 < K; k0 += 32) {
        bf16x16 a0 = load_a_frag(A, K, m0,      k0, lane);
        bf16x16 a1 = load_a_frag(A, K, m0 + 16, k0, lane);
        bf16x16 b0 = load_b_frag(W, K, n0,      k0, lane);
        bf16x16 b1 = load_b_frag(W, K, n0 + 16, k0, lane);
        c00 = WMMA_BF16(a0, b0, c00);
        c01 = WMMA_BF16(a0, b1, c01);
        c10 = WMMA_BF16(a1, b0, c10);
        c11 = WMMA_BF16(a1, b1, c11);
    }
    int cl = lane & 15, half = lane >> 4;
    f32x8 cc[2][2] = {{c00, c01}, {c10, c11}};
#pragma unroll
    for (int mi = 0; mi < 2; mi++) {
#pragma unroll
        for (int ni = 0; ni < 2; ni++) {
            int col = n0 + ni * 16 + cl;
            float bv = bias ? bias[col] : 0.f;
#pragma unroll
            for (int r = 0; r < 8; r++) {
                float v = cc[mi][ni][r] + bv;
                if (relu) v = fmaxf(v, 0.f);
                long idx = (long)(m0 + mi * 16 + half * 8 + r) * N + col;
                if (Cf) Cf[idx] = v;
                if (Cb) Cb[idx] = (bf16)v;
            }
        }
    }
}

// ---------------------------------------------------------------------------
// Batched attention scores: S[b,h,t,s] = scale * sum_d Q[b,t,h*dh+d]*Kk[b,s,h*dh+d]
// (also used for pointer logits with H=1, dh=D). Optional causal mask (s>t).
// One wave per 32x32 block of (t,s), 2x2 register-blocked. T%32==0, S%32==0.
// ---------------------------------------------------------------------------
__global__ void attn_scores_kernel(const bf16* __restrict__ Q, const bf16* __restrict__ Kk,
                                   float* __restrict__ Sout,
                                   int B, int H, int T, int S, int D, int dh,
                                   float scale, int causal) {
    int wave = threadIdx.x >> 5, lane = threadIdx.x & 31;
    int tilesT = T >> 5, tilesS = S >> 5, per = tilesT * tilesS;
    int tile = blockIdx.x * (blockDim.x >> 5) + wave;
    if (tile >= B * H * per) return;
    int bh = tile / per, r = tile % per;
    int t0 = (r / tilesS) << 5, s0 = (r % tilesS) << 5;
    int b = bh / H, h = bh % H;
    const bf16* Qb = Q + (long)b * T * D + h * dh;
    const bf16* Kb = Kk + (long)b * S * D + h * dh;
    f32x8 c00 = {}, c01 = {}, c10 = {}, c11 = {};
    for (int k0 = 0; k0 < dh; k0 += 32) {
        bf16x16 a0 = load_a_frag(Qb, D, t0,      k0, lane);
        bf16x16 a1 = load_a_frag(Qb, D, t0 + 16, k0, lane);
        bf16x16 b0 = load_b_frag(Kb, D, s0,      k0, lane);
        bf16x16 b1 = load_b_frag(Kb, D, s0 + 16, k0, lane);
        c00 = WMMA_BF16(a0, b0, c00);
        c01 = WMMA_BF16(a0, b1, c01);
        c10 = WMMA_BF16(a1, b0, c10);
        c11 = WMMA_BF16(a1, b1, c11);
    }
    int cl = lane & 15, half = lane >> 4;
    f32x8 cc[2][2] = {{c00, c01}, {c10, c11}};
    float* orow = Sout + (long)bh * T * S;
#pragma unroll
    for (int mi = 0; mi < 2; mi++) {
#pragma unroll
        for (int ni = 0; ni < 2; ni++) {
            int col = s0 + ni * 16 + cl;
#pragma unroll
            for (int r8 = 0; r8 < 8; r8++) {
                int row = t0 + mi * 16 + half * 8 + r8;
                float v = cc[mi][ni][r8] * scale;
                if (causal && col > row) v = -1e30f;
                orow[(long)row * S + col] = v;
            }
        }
    }
}

// ---------------------------------------------------------------------------
// Row softmax (S==128), writes bf16 probabilities. One wave per row.
// ---------------------------------------------------------------------------
__global__ void softmax_kernel(const float* __restrict__ X, bf16* __restrict__ P,
                               int rows, int S) {
    int row = blockIdx.x * (blockDim.x >> 5) + (threadIdx.x >> 5);
    int lane = threadIdx.x & 31;
    if (row >= rows) return;
    const float* x = X + (long)row * S;
    float v[4];
    float mx = -3.0e38f;
#pragma unroll
    for (int i = 0; i < 4; i++) { v[i] = x[i * 32 + lane]; mx = fmaxf(mx, v[i]); }
    mx = wave_max(mx);
    float s = 0.f;
#pragma unroll
    for (int i = 0; i < 4; i++) { v[i] = __expf(v[i] - mx); s += v[i]; }
    s = wave_sum(s);
    float inv = 1.f / s;
    bf16* o = P + (long)row * S;
#pragma unroll
    for (int i = 0; i < 4; i++) o[i * 32 + lane] = (bf16)(v[i] * inv);
}

// ---------------------------------------------------------------------------
// O[b,t,h*dh+d] = sum_s P[b,h,t,s] * V[b,s,h*dh+d]
// One wave per 16(t) x 32(=dh) block: one A-frag feeds both d-subtiles.
// ---------------------------------------------------------------------------
__global__ void attn_out_kernel(const bf16* __restrict__ P, const bf16* __restrict__ V,
                                bf16* __restrict__ Ob,
                                int B, int H, int T, int S, int dh, int D) {
    int wave = threadIdx.x >> 5, lane = threadIdx.x & 31;
    int tilesT = T >> 4;
    int tile = blockIdx.x * (blockDim.x >> 5) + wave;
    if (tile >= B * H * tilesT) return;
    int bh = tile / tilesT;
    int t0 = (tile % tilesT) << 4;
    int b = bh / H, h = bh % H;
    const bf16* Pb = P + (long)bh * T * S;
    const bf16* Vc0 = V + (long)b * S * D + h * dh + (lane & 15);
    const bf16* Vc1 = Vc0 + 16;
    int half = lane >> 4;
    f32x8 c0 = {}, c1 = {};
    for (int k0 = 0; k0 < S; k0 += 32) {
        bf16x16 a = load_a_frag(Pb, S, t0, k0, lane);
        bf16x16 b0, b1;
        int ks = k0 + half * 16;
#pragma unroll
        for (int i = 0; i < 16; i++) {
            b0[i] = Vc0[(long)(ks + i) * D];
            b1[i] = Vc1[(long)(ks + i) * D];
        }
        c0 = WMMA_BF16(a, b0, c0);
        c1 = WMMA_BF16(a, b1, c1);
    }
    int col0 = h * dh + (lane & 15);
#pragma unroll
    for (int r8 = 0; r8 < 8; r8++) {
        int row = t0 + half * 8 + r8;
        long base = ((long)b * T + row) * D;
        Ob[base + col0]      = (bf16)c0[r8];
        Ob[base + col0 + 16] = (bf16)c1[r8];
    }
}

// ---------------------------------------------------------------------------
// y = LayerNorm(x + r) * g + b  (D==256, wave per row), writes f32 + bf16
// ---------------------------------------------------------------------------
__global__ void add_ln_kernel(const float* __restrict__ X, const float* __restrict__ R,
                              const float* __restrict__ G, const float* __restrict__ Bt,
                              float* __restrict__ Yf, bf16* __restrict__ Yb,
                              int rows, int D) {
    int row = blockIdx.x * (blockDim.x >> 5) + (threadIdx.x >> 5);
    int lane = threadIdx.x & 31;
    if (row >= rows) return;
    const float* x = X + (long)row * D;
    const float* r = R + (long)row * D;
    float v[8], s = 0.f;
#pragma unroll
    for (int i = 0; i < 8; i++) { int c = i * 32 + lane; v[i] = x[c] + r[c]; s += v[i]; }
    float mu = wave_sum(s) / (float)D;
    float var = 0.f;
#pragma unroll
    for (int i = 0; i < 8; i++) { float d = v[i] - mu; var += d * d; }
    var = wave_sum(var) / (float)D;
    float inv = rsqrtf(var + 1e-5f);
#pragma unroll
    for (int i = 0; i < 8; i++) {
        int c = i * 32 + lane;
        float y = (v[i] - mu) * inv * G[c] + Bt[c];
        Yf[(long)row * D + c] = y;
        Yb[(long)row * D + c] = (bf16)y;
    }
}

// ---------------------------------------------------------------------------
// h = x @ Winp^T + b + pos_enc(N, D)   (Din=16 -> scalar f32, trivial FLOPs)
// ---------------------------------------------------------------------------
__global__ void embed_kernel(const float* __restrict__ x, const float* __restrict__ W,
                             const float* __restrict__ bias,
                             float* __restrict__ hf, bf16* __restrict__ hb,
                             int BN, int N, int Din, int D) {
    long i = (long)blockIdx.x * blockDim.x + threadIdx.x;
    if (i >= (long)BN * D) return;
    int j = (int)(i % D);
    long row = i / D;
    int t = (int)(row % N);
    const float* xr = x + row * Din;
    const float* wr = W + (long)j * Din;
    float acc = bias[j];
#pragma unroll
    for (int k = 0; k < 16; k++) acc += xr[k] * wr[k];
    int jj = j & ~1;
    float ang = (float)t * __expf(-logf(10000.f) * (float)jj / (float)D);
    acc += (j & 1) ? __cosf(ang) : __sinf(ang);
    hf[i] = acc;
    hb[i] = (bf16)acc;
}

// ---------------------------------------------------------------------------
// tgt[b,0]=query_start ; tgt[b,t]=H[b, target_idx[b,t-1]]  (t>=1)
// ---------------------------------------------------------------------------
__global__ void build_tgt_kernel(const float* __restrict__ Hf, const int* __restrict__ tidx,
                                 const float* __restrict__ qs,
                                 float* __restrict__ tf, bf16* __restrict__ tb,
                                 int B, int N, int D) {
    long i = (long)blockIdx.x * blockDim.x + threadIdx.x;
    if (i >= (long)B * N * D) return;
    int j = (int)(i % D);
    long row = i / D;
    int t = (int)(row % N), b = (int)(row / N);
    float v;
    if (t == 0) v = qs[j];
    else {
        int src = tidx[b * N + t - 1];
        v = Hf[((long)b * N + src) * D + j];
    }
    tf[i] = v;
    tb[i] = (bf16)v;
}

// ---------------------------------------------------------------------------
// bias[b,j] = clip( mean_i ( w2 . relu(W1 e[b,i,j,:] + b1) + b2 ), -10, 10 )
// One block per (b,j), 128 threads over i.
// ---------------------------------------------------------------------------
__global__ void edge_bias_kernel(const float* __restrict__ E, const float* __restrict__ w1,
                                 const float* __restrict__ b1, const float* __restrict__ w2,
                                 const float* __restrict__ b2, float* __restrict__ bias,
                                 int N) {
    __shared__ float red[128];
    int bj = blockIdx.x;
    int b = bj / N, j = bj % N;
    int i = threadIdx.x;
    const float* e = E + (((long)b * N + i) * N + j) * 3;
    float e0 = e[0], e1 = e[1], e2 = e[2];
    float acc = b2[0];
#pragma unroll
    for (int k = 0; k < 32; k++) {
        float h = w1[k * 3 + 0] * e0 + w1[k * 3 + 1] * e1 + w1[k * 3 + 2] * e2 + b1[k];
        acc += w2[k] * fmaxf(h, 0.f);
    }
    red[i] = acc;
    __syncthreads();
    for (int s = 64; s > 0; s >>= 1) { if (i < s) red[i] += red[i + s]; __syncthreads(); }
    if (i == 0) {
        float m = red[0] / (float)N;
        bias[bj] = fminf(fmaxf(m, -10.f), 10.f);
    }
}

// ---------------------------------------------------------------------------
// Exclusive cumulative one-hot visited bitmask: vis[(b*N+t)*4 + w]
// ---------------------------------------------------------------------------
__global__ void visited_kernel(const int* __restrict__ tidx, unsigned* __restrict__ vis,
                               int B, int N) {
    int b = blockIdx.x * blockDim.x + threadIdx.x;
    if (b >= B) return;
    unsigned m[4] = {0u, 0u, 0u, 0u};
    for (int t = 0; t < N; t++) {
        long base = ((long)b * N + t) * 4;
        vis[base + 0] = m[0]; vis[base + 1] = m[1];
        vis[base + 2] = m[2]; vis[base + 3] = m[3];
        int id = tidx[b * N + t] & 127;
        m[id >> 5] |= (1u << (id & 31));
    }
}

__global__ void zero_kernel(float* __restrict__ o) { o[0] = 0.f; }

// ---------------------------------------------------------------------------
// Loss: clip(logits+bias,-50,50); visited->-1e4; log_softmax; label smoothing.
// One wave per (b,t) row; atomic accumulate mean.
// ---------------------------------------------------------------------------
__global__ void loss_kernel(const float* __restrict__ logits, const float* __restrict__ bias,
                            const unsigned* __restrict__ vis, const int* __restrict__ tidx,
                            float* __restrict__ out, int B, int N, float invBN) {
    int row = blockIdx.x * (blockDim.x >> 5) + (threadIdx.x >> 5);
    int lane = threadIdx.x & 31;
    if (row >= B * N) return;
    int b = row / N;
    const float* lr = logits + (long)row * N;
    const float* br = bias + (long)b * N;
    const unsigned* vm = vis + (long)row * 4;
    int tgt = tidx[row];
    float vals[4];
    int isv[4];
    float mx = -3.0e38f, unv = 0.f;
#pragma unroll
    for (int i = 0; i < 4; i++) {
        int n = i * 32 + lane;
        float v = fminf(fmaxf(lr[n] + br[n], -50.f), 50.f);
        int m = (vm[n >> 5] >> (n & 31)) & 1;
        if (m) v = -10000.f;
        vals[i] = v; isv[i] = m;
        mx = fmaxf(mx, v);
        unv += m ? 0.f : 1.f;
    }
    mx = wave_max(mx);
    unv = wave_sum(unv);
    float se = 0.f;
#pragma unroll
    for (int i = 0; i < 4; i++) se += __expf(vals[i] - mx);
    se = wave_sum(se);
    float logZ = mx + __logf(se);
    float acc = 0.f;
#pragma unroll
    for (int i = 0; i < 4; i++) {
        int n = i * 32 + lane;
        float sm = (n == tgt ? 0.95f : 0.f) + (isv[i] ? 0.f : 0.05f / unv);
        acc += sm * (vals[i] - logZ);
    }
    acc = wave_sum(acc);
    if (lane == 0) atomicAdd(out, -acc * invBN);
}

// ---------------------------------------------------------------------------
// Host-side parameter bookkeeping (jax tree_leaves order: sorted dict keys)
// ---------------------------------------------------------------------------
struct LinP  { const float* b; const float* w; };
struct AttnP { LinP k, o, q, v; };
struct LnP   { const float* b; const float* g; };
struct DecP  { AttnP ca; LinP ff1, ff2; LnP ln1, ln2, ln3; AttnP sa; };
struct EncP  { LinP ff1, ff2; LnP ln1, ln2; AttnP sa; };
struct AttnW { bf16 *qw, *kw, *vw, *ow; const float *qb, *kb, *vb, *ob; };

extern "C" void kernel_launch(void* const* d_in, const int* in_sizes, int n_in,
                              void* d_out, int out_size, void* d_ws, size_t ws_size,
                              hipStream_t stream) {
    (void)in_sizes; (void)out_size; (void)ws_size;
    const int B = 32, N = 128, Din = 16, D = 256, H = 8, dh = 32, DFF = 1024;
    const int BN = B * N;  // 4096 rows
    if (n_in < 98) return;  // unexpected flattening; avoid OOB host reads

    const float* x    = (const float*)d_in[0];
    const int*   tidx = (const int*)d_in[1];
    const float* edge = (const float*)d_in[2];

    // ---- walk param leaves (sorted-key flatten) ----
    int p = 3;
    auto nextf  = [&]() -> const float* { return (const float*)d_in[p++]; };
    auto rdLin  = [&](LinP& L)  { L.b = nextf(); L.w = nextf(); };
    auto rdAttn = [&](AttnP& A) { rdLin(A.k); rdLin(A.o); rdLin(A.q); rdLin(A.v); };
    auto rdLn   = [&](LnP& L)   { L.b = nextf(); L.g = nextf(); };

    DecP dec[2];
    for (int l = 0; l < 2; l++) {
        rdAttn(dec[l].ca); rdLin(dec[l].ff1); rdLin(dec[l].ff2);
        rdLn(dec[l].ln1); rdLn(dec[l].ln2); rdLn(dec[l].ln3); rdAttn(dec[l].sa);
    }
    LinP eb1, eb2; rdLin(eb1); rdLin(eb2);
    EncP enc[2];
    for (int l = 0; l < 2; l++) {
        rdLin(enc[l].ff1); rdLin(enc[l].ff2);
        rdLn(enc[l].ln1); rdLn(enc[l].ln2); rdAttn(enc[l].sa);
    }
    LinP inp;  rdLin(inp);
    LinP outp; rdLin(outp);
    LinP ptrp; rdLin(ptrp);
    const float* qstart = nextf();

    // ---- workspace bump allocator ----
    char* base = (char*)d_ws;
    size_t off = 0;
    auto alloc  = [&](size_t bytes) -> void* {
        off = (off + 255) & ~(size_t)255;
        void* r = base + off; off += bytes; return r;
    };
    auto allocF = [&](size_t n) { return (float*)alloc(n * 4); };
    auto allocB = [&](size_t n) { return (bf16*)alloc(n * 2); };

    auto cvt = [&](const float* src, size_t n) -> bf16* {
        bf16* d = allocB(n);
        cvt_kernel<<<(int)((n + 255) / 256), 256, 0, stream>>>(src, d, (long)n);
        return d;
    };
    auto mkAttnW = [&](const AttnP& a) -> AttnW {
        AttnW w;
        w.qw = cvt(a.q.w, (size_t)D * D); w.kw = cvt(a.k.w, (size_t)D * D);
        w.vw = cvt(a.v.w, (size_t)D * D); w.ow = cvt(a.o.w, (size_t)D * D);
        w.qb = a.q.b; w.kb = a.k.b; w.vb = a.v.b; w.ob = a.o.b;
        return w;
    };

    // ---- convert GEMM weights to bf16 ----
    AttnW encSA[2], decSA[2], decCA[2];
    bf16 *encF1[2], *encF2[2], *decF1[2], *decF2[2];
    for (int l = 0; l < 2; l++) {
        encSA[l] = mkAttnW(enc[l].sa);
        encF1[l] = cvt(enc[l].ff1.w, (size_t)DFF * D);
        encF2[l] = cvt(enc[l].ff2.w, (size_t)D * DFF);
    }
    for (int l = 0; l < 2; l++) {
        decSA[l] = mkAttnW(dec[l].sa);
        decCA[l] = mkAttnW(dec[l].ca);
        decF1[l] = cvt(dec[l].ff1.w, (size_t)DFF * D);
        decF2[l] = cvt(dec[l].ff2.w, (size_t)D * DFF);
    }
    bf16* outW = cvt(outp.w, (size_t)D * D);
    bf16* ptrW = cvt(ptrp.w, (size_t)D * D);

    // ---- activation buffers ----
    float* a0f = allocF((size_t)BN * D); bf16* a0b = allocB((size_t)BN * D);
    float* a1f = allocF((size_t)BN * D); bf16* a1b = allocB((size_t)BN * D);
    float* d0f = allocF((size_t)BN * D); bf16* d0b = allocB((size_t)BN * D);
    float* d1f = allocF((size_t)BN * D); bf16* d1b = allocB((size_t)BN * D);
    bf16* Qb = allocB((size_t)BN * D);
    bf16* Kb = allocB((size_t)BN * D);
    bf16* Vb = allocB((size_t)BN * D);
    float* scoresF = allocF((size_t)B * H * N * N);
    bf16*  probsB  = allocB((size_t)B * H * N * N);
    bf16*  AOb     = allocB((size_t)BN * D);
    float* projF   = allocF((size_t)BN * D);
    bf16*  midB    = allocB((size_t)BN * DFF);
    float* ffoutF  = allocF((size_t)BN * D);
    bf16*  qptrB   = allocB((size_t)BN * D);
    bf16*  keysB   = allocB((size_t)BN * D);
    float* logitsF = allocF((size_t)BN * N);
    float* biasF   = allocF((size_t)B * N);
    unsigned* visU = (unsigned*)alloc((size_t)BN * 4 * sizeof(unsigned));

    // ---- launch helpers ----
    auto gemm = [&](const bf16* A, const bf16* W, const float* bias,
                    float* Cf, bf16* Cb, int M, int Nc, int K, int relu) {
        int tiles = (M / 32) * (Nc / 32);
        gemm_bf16_kernel<<<(tiles + 7) / 8, 256, 0, stream>>>(A, W, bias, Cf, Cb, M, Nc, K, relu);
    };
    auto run_mha = [&](const bf16* qinB, const bf16* kvinB, const AttnW& aw, int causal) {
        gemm(qinB,  aw.qw, aw.qb, nullptr, Qb, BN, D, D, 0);
        gemm(kvinB, aw.kw, aw.kb, nullptr, Kb, BN, D, D, 0);
        gemm(kvinB, aw.vw, aw.vb, nullptr, Vb, BN, D, D, 0);
        int stiles = B * H * (N / 32) * (N / 32);
        attn_scores_kernel<<<(stiles + 7) / 8, 256, 0, stream>>>(
            Qb, Kb, scoresF, B, H, N, N, D, dh, 0.17677669529663687f /*1/sqrt(32)*/, causal);
        int rows = B * H * N;
        softmax_kernel<<<(rows + 7) / 8, 256, 0, stream>>>(scoresF, probsB, rows, N);
        int otiles = B * H * (N / 16);
        attn_out_kernel<<<(otiles + 7) / 8, 256, 0, stream>>>(probsB, Vb, AOb, B, H, N, N, dh, D);
        gemm(AOb, aw.ow, aw.ob, projF, nullptr, BN, D, D, 0);
    };
    auto run_ln = [&](const float* xres, const float* delta, const LnP& l, float* yf, bf16* yb) {
        add_ln_kernel<<<(BN + 7) / 8, 256, 0, stream>>>(xres, delta, l.g, l.b, yf, yb, BN, D);
    };

    // ================= Encoder =================
    embed_kernel<<<(int)(((long)BN * D + 255) / 256), 256, 0, stream>>>(
        x, inp.w, inp.b, a0f, a0b, BN, N, Din, D);
    float* curF = a0f; bf16* curB = a0b;
    float* altF = a1f; bf16* altB = a1b;
    for (int l = 0; l < 2; l++) {
        run_mha(curB, curB, encSA[l], 0);
        run_ln(curF, projF, enc[l].ln1, altF, altB);
        { float* tf = curF; curF = altF; altF = tf; bf16* tb = curB; curB = altB; altB = tb; }
        gemm(curB, encF1[l], enc[l].ff1.b, nullptr, midB, BN, DFF, D, 1);
        gemm(midB, encF2[l], enc[l].ff2.b, ffoutF, nullptr, BN, D, DFF, 0);
        run_ln(curF, ffoutF, enc[l].ln2, altF, altB);
        { float* tf = curF; curF = altF; altF = tf; bf16* tb = curB; curB = altB; altB = tb; }
    }
    float* Hf = curF; bf16* Hb = curB;  // encoder memory (kept intact)

    // ================= Decoder =================
    build_tgt_kernel<<<(int)(((long)BN * D + 255) / 256), 256, 0, stream>>>(
        Hf, tidx, qstart, d0f, d0b, B, N, D);
    curF = d0f; curB = d0b; altF = d1f; altB = d1b;
    for (int l = 0; l < 2; l++) {
        run_mha(curB, curB, decSA[l], 1);                 // causal self-attention
        run_ln(curF, projF, dec[l].ln1, altF, altB);
        { float* tf = curF; curF = altF; altF = tf; bf16* tb = curB; curB = altB; altB = tb; }
        run_mha(curB, Hb, decCA[l], 0);                   // cross-attention to encoder
        run_ln(curF, projF, dec[l].ln2, altF, altB);
        { float* tf = curF; curF = altF; altF = tf; bf16* tb = curB; curB = altB; altB = tb; }
        gemm(curB, decF1[l], dec[l].ff1.b, nullptr, midB, BN, DFF, D, 1);
        gemm(midB, decF2[l], dec[l].ff2.b, ffoutF, nullptr, BN, D, DFF, 0);
        run_ln(curF, ffoutF, dec[l].ln3, altF, altB);
        { float* tf = curF; curF = altF; altF = tf; bf16* tb = curB; curB = altB; altB = tb; }
    }

    // ================= Pointer logits + loss =================
    gemm(curB, outW, outp.b, nullptr, qptrB, BN, D, D, 0);
    gemm(Hb,  ptrW, ptrp.b, nullptr, keysB, BN, D, D, 0);
    int ltiles = B * (N / 32) * (N / 32);
    attn_scores_kernel<<<(ltiles + 7) / 8, 256, 0, stream>>>(
        qptrB, keysB, logitsF, B, 1, N, N, D, D, 0.0625f /*1/sqrt(256)*/, 0);

    edge_bias_kernel<<<B * N, 128, 0, stream>>>(edge, eb1.w, eb1.b, eb2.w, eb2.b, biasF, N);
    visited_kernel<<<1, 32, 0, stream>>>(tidx, visU, B, N);
    zero_kernel<<<1, 1, 0, stream>>>((float*)d_out);
    loss_kernel<<<(BN + 7) / 8, 256, 0, stream>>>(
        logitsF, biasF, visU, tidx, (float*)d_out, B, N, 1.f / (float)BN);
}
